// MaskedSelfAttention_33097017983494
// MI455X (gfx1250) — compile-verified, matrix operand order checked by reading, not run on hardware
//
#include <hip/hip_runtime.h>

typedef unsigned short u16;
typedef unsigned int   u32;

typedef __attribute__((ext_vector_type(16))) __bf16 v16bf;
typedef __attribute__((ext_vector_type(8)))  float  v8f;

// ---------------------------------------------------------------------------
// Shapes (hardcoded from reference): B=4, S=2048, D=1024, H=16, DK=64
// ---------------------------------------------------------------------------
#define SB   4
#define SS   2048
#define SD   1024
#define SH   16
#define SDK  64
#define SM   (SB * SS)

union FragU {
  v16bf v;
  uint4 q[2];
};

__device__ __forceinline__ u16 f2bf(float f) {
  u32 u = __float_as_uint(f);
  u32 r = u + 0x7FFFu + ((u >> 16) & 1u);   // round-to-nearest-even
  return (u16)(r >> 16);
}

__device__ __forceinline__ v8f vzero8() {
  v8f z;
#pragma unroll
  for (int i = 0; i < 8; ++i) z[i] = 0.0f;
  return z;
}

// Low 32 bits of a flat shared pointer = LDS byte address (ISA 10.2 aperture).
__device__ __forceinline__ u32 lds_off32(const void* p) {
  return (u32)(size_t)p;
}

// Async copy of 32 bytes (two b128) global -> LDS. INST_OFFSET is added to
// both the global and LDS addresses per the async-load pseudocode, so one
// address pair serves both halves. Tracked by ASYNCcnt.
__device__ __forceinline__ void async_copy_b128x2(u32 lds, const u16* g) {
  unsigned long long ga = (unsigned long long)(size_t)g;
  asm volatile("global_load_async_to_lds_b128 %0, %1, off"
               :: "v"(lds), "v"(ga) : "memory");
  asm volatile("global_load_async_to_lds_b128 %0, %1, off offset:16"
               :: "v"(lds), "v"(ga) : "memory");
}

__device__ __forceinline__ void wait_async0() {
  asm volatile("s_wait_asynccnt 0x0" ::: "memory");
}

// Load a 16x32 bf16 A/B fragment (16-bit WMMA operand layout) from a
// K-contiguous [16 x stride] tile. Lanes 0-15 hold K 0-7 & 16-23, lanes 16-31
// hold K 8-15 & 24-31, row = lane%16. Two b128 loads.
__device__ __forceinline__ v16bf load_frag(const u16* base, int stride, int lane) {
  const int row  = lane & 15;
  const int koff = (lane >> 4) * 8;
  const u16* p = base + row * stride + koff;
  FragU f;
  f.q[0] = *(const uint4*)(p);
  f.q[1] = *(const uint4*)(p + 16);
  return f.v;
}

__device__ __forceinline__ v8f wmma_bf16(v16bf a, v16bf b, v8f c) {
  return __builtin_amdgcn_wmma_f32_16x16x32_bf16(
      /*neg_a=*/false, a, /*neg_b=*/false, b,
      /*c_mod=*/(short)0, c, /*reuse_a=*/false, /*reuse_b=*/false);
}

__device__ __forceinline__ float hmax16(float v) {
  v = fmaxf(v, __shfl_xor(v, 1, 32));
  v = fmaxf(v, __shfl_xor(v, 2, 32));
  v = fmaxf(v, __shfl_xor(v, 4, 32));
  v = fmaxf(v, __shfl_xor(v, 8, 32));
  return v;
}
__device__ __forceinline__ float hsum16(float v) {
  v += __shfl_xor(v, 1, 32);
  v += __shfl_xor(v, 2, 32);
  v += __shfl_xor(v, 4, 32);
  v += __shfl_xor(v, 8, 32);
  return v;
}

// ---------------------------------------------------------------------------
// fp32 -> bf16 conversion, 4 elements/thread
// ---------------------------------------------------------------------------
__global__ void __launch_bounds__(256)
cvt_f32_bf16_kernel(const float4* __restrict__ in, uint2* __restrict__ out, int n4) {
  int i = blockIdx.x * blockDim.x + threadIdx.x;
  if (i >= n4) return;
  float4 f = in[i];
  uint2 r;
  r.x = (u32)f2bf(f.x) | ((u32)f2bf(f.y) << 16);
  r.y = (u32)f2bf(f.z) | ((u32)f2bf(f.w) << 16);
  out[i] = r;
}

// ---------------------------------------------------------------------------
// WMMA GEMM:  C[M,N] = A[M,K] * Bm[N,K]^T   (torch Linear semantics)
// A, Bm bf16, K = N = 1024, M = 8192. Block tile 128x128, 8 waves of 32x64.
// Double-buffered LDS with async global->LDS staging: loads for tile t+1 are
// issued right after the barrier and overlap the 8 WMMAs on tile t.
// mode 0: write bf16 into [B,H,S,DK]      (Q, K projections)
// mode 1: write bf16 into [B,H,DK,S]      (V projection, pre-transposed)
// mode 2: write fp32 row-major + bias     (output projection)
// ---------------------------------------------------------------------------
#define GSTRIDE 40   // padded LDS K-stride (bf16 elems), multiple of 8

__global__ void __launch_bounds__(256)
gemm_wmma_kernel(const u16* __restrict__ A, const u16* __restrict__ Bm,
                 float* __restrict__ outF, u16* __restrict__ outB,
                 const float* __restrict__ bias, int mode) {
  __shared__ u16 ldsA[2][128 * GSTRIDE];
  __shared__ u16 ldsB[2][128 * GSTRIDE];

  const int tid  = threadIdx.x;
  const int lane = tid & 31;
  const int wave = tid >> 5;
  const int hi   = lane >> 4;
  const int ln   = lane & 15;
  const int wm   = wave >> 1;   // 0..3  (M direction, 32 rows each)
  const int wn   = wave & 1;    // 0..1  (N direction, 64 cols each)
  const int mb   = blockIdx.x * 128;
  const int nb   = blockIdx.y * 128;

  v8f acc[2][4];
#pragma unroll
  for (int mf = 0; mf < 2; ++mf)
#pragma unroll
    for (int nf = 0; nf < 4; ++nf) acc[mf][nf] = vzero8();

  // staging role: each thread owns 16 contiguous bf16 (two b128) per matrix
  const int lrow = tid >> 1;          // 0..127
  const int lcol = (tid & 1) * 16;    // 0 or 16
  const u16* agBase = A  + (size_t)(mb + lrow) * SD + lcol;
  const u16* bgBase = Bm + (size_t)(nb + lrow) * SD + lcol;
  u32 laA[2], laB[2];
  laA[0] = lds_off32(&ldsA[0][lrow * GSTRIDE + lcol]);
  laA[1] = lds_off32(&ldsA[1][lrow * GSTRIDE + lcol]);
  laB[0] = lds_off32(&ldsB[0][lrow * GSTRIDE + lcol]);
  laB[1] = lds_off32(&ldsB[1][lrow * GSTRIDE + lcol]);

  // prologue: stage tile 0 into buffer 0
  async_copy_b128x2(laA[0], agBase);
  async_copy_b128x2(laB[0], bgBase);

  const int nT = SD / 32;
  for (int it = 0; it < nT; ++it) {
    wait_async0();        // this wave's async writes for tile `it` landed
    __syncthreads();      // all waves' writes visible; prior reads of buf^1 done
    if (it + 1 < nT) {    // stage tile it+1; overlaps with compute below
      async_copy_b128x2(laA[(it + 1) & 1], agBase + (it + 1) * 32);
      async_copy_b128x2(laB[(it + 1) & 1], bgBase + (it + 1) * 32);
    }

    const u16* bufA = ldsA[it & 1];
    const u16* bufB = ldsB[it & 1];
    v16bf af0 = load_frag(bufA + (wm * 32)      * GSTRIDE, GSTRIDE, lane);
    v16bf af1 = load_frag(bufA + (wm * 32 + 16) * GSTRIDE, GSTRIDE, lane);
#pragma unroll
    for (int nf = 0; nf < 4; ++nf) {
      v16bf bf = load_frag(bufB + (wn * 64 + nf * 16) * GSTRIDE, GSTRIDE, lane);
      acc[0][nf] = wmma_bf16(af0, bf, acc[0][nf]);
      acc[1][nf] = wmma_bf16(af1, bf, acc[1][nf]);
    }
  }

  // Store. C-fragment layout: vgpr r -> row (r + 8*hi), lane%16 -> col.
#pragma unroll
  for (int mf = 0; mf < 2; ++mf)
#pragma unroll
    for (int nf = 0; nf < 4; ++nf)
#pragma unroll
      for (int r = 0; r < 8; ++r) {
        const int m = mb + wm * 32 + mf * 16 + r + hi * 8;
        const int n = nb + wn * 64 + nf * 16 + ln;
        const float v = acc[mf][nf][r];
        if (mode == 2) {
          outF[(size_t)m * SD + n] = v + bias[n];
        } else {
          const int b = m >> 11, s = m & (SS - 1);
          const int h = n >> 6,  dk = n & (SDK - 1);
          if (mode == 0)
            outB[(((size_t)(b * SH + h)) * SS + s) * SDK + dk] = f2bf(v);
          else
            outB[(((size_t)(b * SH + h)) * SDK + dk) * SS + s] = f2bf(v);
        }
      }
}

// ---------------------------------------------------------------------------
// Flash attention (causal). Grid: (S/128, B*H). 256 threads = 8 waves.
// Wave w owns 16 query rows. K/V 64-key blocks async-staged in LDS by the
// whole block; P (16x64 per wave) bounced through private LDS to re-layout
// C-fragment -> A-fragment for the P@V WMMA.
// Q,K in [B,H,S,DK] bf16; V in [B,H,DK,S] bf16; ctx out [B,S,D] bf16.
// ---------------------------------------------------------------------------
#define FSTRIDE 72   // padded LDS stride (bf16 elems), multiple of 8

__global__ void __launch_bounds__(256)
flash_attn_kernel(const u16* __restrict__ Q, const u16* __restrict__ K,
                  const u16* __restrict__ Vt, u16* __restrict__ ctx) {
  __shared__ u16 ldsK[64 * FSTRIDE];
  __shared__ u16 ldsV[64 * FSTRIDE];
  __shared__ u16 ldsP[8][16 * FSTRIDE];

  const int tid  = threadIdx.x;
  const int lane = tid & 31;
  const int wave = tid >> 5;
  const int hi   = lane >> 4;
  const int ln   = lane & 15;
  const int q0   = blockIdx.x * 128;
  const int bh   = blockIdx.y;          // b*H + h
  const int qw   = q0 + wave * 16;      // wave's first query row

  // Q fragments (A layout), two 32-wide K-chunks over d = 0..63
  v16bf qf[2];
  {
    const u16* qbase = Q + ((size_t)bh * SS + qw) * SDK;
#pragma unroll
    for (int c = 0; c < 2; ++c) {
      const u16* p = qbase + (size_t)ln * SDK + c * 32 + hi * 8;
      FragU f;
      f.q[0] = *(const uint4*)(p);
      f.q[1] = *(const uint4*)(p + 16);
      qf[c] = f.v;
    }
  }

  v8f acc[4];
#pragma unroll
  for (int j = 0; j < 4; ++j) acc[j] = vzero8();
  float mrow[8], lrowv[8];
#pragma unroll
  for (int r = 0; r < 8; ++r) { mrow[r] = -3.0e38f; lrowv[r] = 0.0f; }

  // staging role: thread owns 16 contiguous bf16 (two b128) per matrix
  const int lrow = tid >> 2;          // 0..63
  const int lcol = (tid & 3) * 16;    // 0,16,32,48
  const u32 laK = lds_off32(&ldsK[lrow * FSTRIDE + lcol]);
  const u32 laV = lds_off32(&ldsV[lrow * FSTRIDE + lcol]);
  const u16* kgBase = K  + ((size_t)bh * SS + lrow) * SDK + lcol;   // + kb*SDK
  const u16* vgBase = Vt + ((size_t)bh * SDK + lrow) * SS + lcol;   // + kb

  const int kend = q0 + 128;          // causal: keys beyond query tile unused

  for (int kb = 0; kb < kend; kb += 64) {
    __syncthreads();                            // previous block's reads done
    async_copy_b128x2(laK, kgBase + (size_t)kb * SDK);
    async_copy_b128x2(laV, vgBase + kb);
    wait_async0();
    __syncthreads();

    // S = Q @ K^T  (16 q-rows x 64 keys per wave)
    v8f sf[4];
#pragma unroll
    for (int j = 0; j < 4; ++j) {
      v8f s = vzero8();
      s = wmma_bf16(qf[0], load_frag(ldsK + j * 16 * FSTRIDE,      FSTRIDE, lane), s);
      s = wmma_bf16(qf[1], load_frag(ldsK + j * 16 * FSTRIDE + 32, FSTRIDE, lane), s);
      sf[j] = s;
    }

    // scale + causal mask + block row-max
    float mblk[8];
#pragma unroll
    for (int r = 0; r < 8; ++r) {
      const int q_idx = qw + r + hi * 8;
      float mx = -3.0e38f;
#pragma unroll
      for (int j = 0; j < 4; ++j) {
        const int k_idx = kb + j * 16 + ln;
        float v = sf[j][r] * 0.125f;              // 1/sqrt(DK)
        v = (k_idx > q_idx) ? -1.0e30f : v;
        sf[j][r] = v;
        mx = fmaxf(mx, v);
      }
      mblk[r] = hmax16(mx);
    }

    // online softmax update; write P (bf16) to per-wave LDS in row-major
#pragma unroll
    for (int r = 0; r < 8; ++r) {
      const float mnew = fmaxf(mrow[r], mblk[r]);
      const float corr = __expf(mrow[r] - mnew);
      mrow[r] = mnew;
      float rs = 0.0f;
#pragma unroll
      for (int j = 0; j < 4; ++j) {
        const float p = __expf(sf[j][r] - mnew);
        ldsP[wave][(r + hi * 8) * FSTRIDE + j * 16 + ln] = f2bf(p);
        rs += p;
      }
      lrowv[r] = lrowv[r] * corr + hsum16(rs);
#pragma unroll
      for (int j = 0; j < 4; ++j) acc[j][r] *= corr;
    }

    // ctx += P @ V  (two 32-wide key chunks)
#pragma unroll
    for (int c = 0; c < 2; ++c) {
      v16bf pf = load_frag(&ldsP[wave][c * 32], FSTRIDE, lane);
#pragma unroll
      for (int j = 0; j < 4; ++j)
        acc[j] = wmma_bf16(pf, load_frag(ldsV + j * 16 * FSTRIDE + c * 32, FSTRIDE, lane), acc[j]);
    }
  }

  // normalize and write ctx to [B,S,D] bf16 (D col = h*64 + d)
  const int b = bh >> 4, h = bh & (SH - 1);
#pragma unroll
  for (int j = 0; j < 4; ++j)
#pragma unroll
    for (int r = 0; r < 8; ++r) {
      const int q_idx = qw + r + hi * 8;
      const int d = j * 16 + ln;
      const float v = acc[j][r] / lrowv[r];
      ctx[((size_t)b * SS + q_idx) * SD + h * SDK + d] = f2bf(v);
    }
}

// ---------------------------------------------------------------------------
// Host launcher. Workspace layout (bf16 elems): x | wq wk wv wo | Q K Vt ctx
// Total = 92 MB.
// ---------------------------------------------------------------------------
extern "C" void kernel_launch(void* const* d_in, const int* in_sizes, int n_in,
                              void* d_out, int out_size, void* d_ws, size_t ws_size,
                              hipStream_t stream) {
  (void)in_sizes; (void)n_in; (void)out_size; (void)ws_size;

  const float* x   = (const float*)d_in[0];
  const float* w_q = (const float*)d_in[1];
  const float* w_k = (const float*)d_in[2];
  const float* w_v = (const float*)d_in[3];
  const float* w_o = (const float*)d_in[4];
  const float* b_o = (const float*)d_in[5];
  float* out = (float*)d_out;

  u16* ws  = (u16*)d_ws;
  u16* xb  = ws;                                      // 8192*1024
  u16* wqb = xb  + (size_t)SM * SD;                   // 1024*1024 each
  u16* wkb = wqb + (size_t)SD * SD;
  u16* wvb = wkb + (size_t)SD * SD;
  u16* wob = wvb + (size_t)SD * SD;
  u16* Qb  = wob + (size_t)SD * SD;                   // 8192*1024 each
  u16* Kb  = Qb  + (size_t)SM * SD;
  u16* Vb  = Kb  + (size_t)SM * SD;
  u16* Cb  = Vb  + (size_t)SM * SD;

  // 1) fp32 -> bf16 converts
  {
    int n4 = (SM * SD) / 4;
    cvt_f32_bf16_kernel<<<n4 / 256, 256, 0, stream>>>((const float4*)x, (uint2*)xb, n4);
    int w4 = (SD * SD) / 4;
    cvt_f32_bf16_kernel<<<w4 / 256, 256, 0, stream>>>((const float4*)w_q, (uint2*)wqb, w4);
    cvt_f32_bf16_kernel<<<w4 / 256, 256, 0, stream>>>((const float4*)w_k, (uint2*)wkb, w4);
    cvt_f32_bf16_kernel<<<w4 / 256, 256, 0, stream>>>((const float4*)w_v, (uint2*)wvb, w4);
    cvt_f32_bf16_kernel<<<w4 / 256, 256, 0, stream>>>((const float4*)w_o, (uint2*)wob, w4);
  }

  // 2) Q/K/V projections (WMMA GEMM)
  dim3 gg(SM / 128, SD / 128);
  gemm_wmma_kernel<<<gg, 256, 0, stream>>>(xb, wqb, nullptr, Qb, nullptr, 0);
  gemm_wmma_kernel<<<gg, 256, 0, stream>>>(xb, wkb, nullptr, Kb, nullptr, 0);
  gemm_wmma_kernel<<<gg, 256, 0, stream>>>(xb, wvb, nullptr, Vb, nullptr, 1);

  // 3) causal flash attention
  flash_attn_kernel<<<dim3(SS / 128, SB * SH), 256, 0, stream>>>(Qb, Kb, Vb, Cb);

  // 4) output projection + bias (fp32 out)
  gemm_wmma_kernel<<<gg, 256, 0, stream>>>(Cb, wob, out, nullptr, b_o, 2);
}
